// sLSTMCell_21380347199691
// MI455X (gfx1250) — compile-verified, matrix operand order checked by reading, not run on hardware
//
#include <hip/hip_runtime.h>

// sLSTM cell, fully fused: pre = x@[Wz|Wi|Wf|Wo] + h@[Uz|Ui|Uf|Uo] + b, then
// exponential-gate pointwise chain. GEMMs run on v_wmma_f32_16x16x32_bf16
// using an f32 -> (bf16_hi, bf16_lo) split (3 WMMAs per product: hh + hl + lh)
// to recover ~f32 accuracy on the fast bf16 matrix path.
// Ping-pong LDS double buffering: ONE barrier per K-step; global prefetch for
// step s+1 issues right after the barrier, WMMAs of step s run from buf[s&1],
// and the convert+ds_store of step s+1 into buf[(s+1)&1] overlaps the WMMAs.

#define B_  256
#define D_  2048
#define U_  2048
#define BM  128     // rows of output tile per workgroup
#define BN  32      // cols of output tile per workgroup
#define KB  32      // K-step (matches 16x16x32 bf16 WMMA K)
#define AS  40      // LDS A row stride in bf16 elems (80B rows, conflict-free)
#define BS  40      // LDS B row stride in bf16 elems
#define KSTEPS (D_ / KB)       // 64 per GEMM phase
#define NSTEP  (2 * KSTEPS)    // 128 total (phase 0: x@W, phase 1: h@U)

typedef __bf16 bf16_t;
typedef __attribute__((ext_vector_type(4)))  __bf16 v4bf;
typedef __attribute__((ext_vector_type(8)))  __bf16 v8bf;
typedef __attribute__((ext_vector_type(16))) __bf16 v16bf;
typedef __attribute__((ext_vector_type(8)))  float  v8f;

// Load one 16-bit WMMA A/B fragment from LDS (row-major rows of 32 bf16).
// ISA layout (16-bit A 16x32 / symmetric B): lane<16 -> K {0..7,16..23},
// lane>=16 -> K {8..15,24..31}. Caller passes pointer at row*stride + kbase.
__device__ __forceinline__ v16bf ld_frag(const bf16_t* p) {
    v8bf a = *(const v8bf*)(p);
    v8bf b = *(const v8bf*)(p + 16);
    return __builtin_shufflevector(a, b, 0,1,2,3,4,5,6,7,8,9,10,11,12,13,14,15);
}

__device__ __forceinline__ void split_bf16(float x, bf16_t& h, bf16_t& l) {
    h = (bf16_t)x;
    l = (bf16_t)(x - (float)h);
}

__global__ __launch_bounds__(256) void slstm_wmma_kernel(
    const float* __restrict__ x,     const float* __restrict__ hprev,
    const float* __restrict__ cprev, const float* __restrict__ nprev,
    const float* __restrict__ mprev,
    const float* __restrict__ Wz, const float* __restrict__ Wi,
    const float* __restrict__ Wf, const float* __restrict__ Wo,
    const float* __restrict__ pbz, const float* __restrict__ pbi,
    const float* __restrict__ pbf, const float* __restrict__ pbo,
    const float* __restrict__ Uz, const float* __restrict__ Ui,
    const float* __restrict__ Uf, const float* __restrict__ Uo,
    float* __restrict__ out)
{
    // ping-pong LDS tiles
    __shared__ __align__(16) bf16_t sA_hi[2][BM * AS];
    __shared__ __align__(16) bf16_t sA_lo[2][BM * AS];
    __shared__ __align__(16) bf16_t sB_hi[2][4 * BN * BS];
    __shared__ __align__(16) bf16_t sB_lo[2][4 * BN * BS];

    const int tid = threadIdx.x;
    const int R0  = blockIdx.y * BM;   // batch-row base
    const int C0  = blockIdx.x * BN;   // unit-col base

    // accumulators: 4 gates x 2 M-tiles of 16x16 f32
    v8f acc[4][2];
    {
        v8f z = {};
        #pragma unroll
        for (int g = 0; g < 4; ++g)
            #pragma unroll
            for (int mt = 0; mt < 2; ++mt)
                acc[g][mt] = z;
    }

    const int wave = tid >> 5;
    const int wr   = wave & 3;    // 4 row-groups of 32
    const int wc   = wave >> 2;   // 2 col-groups of 16
    const int lane = tid & 31;
    const int lr   = lane & 15;
    const int hl   = lane >> 4;
    const int kb   = hl * 8;      // fragment K base per half-wave

    // staging roles for B: 64 threads per gate
    const int sg  = tid >> 6;
    const int sub = tid & 63;

    // staging index precompute
    const int arow[4] = { (tid + 0*256) >> 3, (tid + 1*256) >> 3,
                          (tid + 2*256) >> 3, (tid + 3*256) >> 3 };
    const int ac4     = tid & 7;
    const int bk [4]  = { (sub + 0*64) >> 3, (sub + 1*64) >> 3,
                          (sub + 2*64) >> 3, (sub + 3*64) >> 3 };
    const int bn4     = sub & 7;

    // ---- register prefetch buffers ----
    float4 pa[4], pb[4];

    auto prefetch = [&](int s) {
        const int ph = s >> 6;               // phase: 0 = x@W, 1 = h@U
        const int kk = (s & (KSTEPS - 1)) * KB;
        const float* Ag = ph ? hprev : x;
        const float* B0 = ph ? Uz : Wz;
        const float* B1 = ph ? Ui : Wi;
        const float* B2 = ph ? Uf : Wf;
        const float* B3 = ph ? Uo : Wo;
        const float* Bg = (sg == 0) ? B0 : (sg == 1) ? B1 : (sg == 2) ? B2 : B3;
        #pragma unroll
        for (int i = 0; i < 4; ++i)
            pa[i] = *(const float4*)(Ag + (size_t)(R0 + arow[i]) * D_ + kk + ac4 * 4);
        #pragma unroll
        for (int j = 0; j < 4; ++j)
            pb[j] = *(const float4*)(Bg + (size_t)(kk + bk[j]) * U_ + C0 + bn4 * 4);
    };

    auto stage = [&](int buf) {
        bf16_t* dAh = sA_hi[buf];
        bf16_t* dAl = sA_lo[buf];
        bf16_t* dBh = sB_hi[buf];
        bf16_t* dBl = sB_lo[buf];
        #pragma unroll
        for (int i = 0; i < 4; ++i) {
            bf16_t h0,l0,h1,l1,h2,l2,h3,l3;
            split_bf16(pa[i].x, h0, l0); split_bf16(pa[i].y, h1, l1);
            split_bf16(pa[i].z, h2, l2); split_bf16(pa[i].w, h3, l3);
            v4bf vh, vl;
            vh[0]=h0; vh[1]=h1; vh[2]=h2; vh[3]=h3;
            vl[0]=l0; vl[1]=l1; vl[2]=l2; vl[3]=l3;
            *(v4bf*)(&dAh[arow[i] * AS + ac4 * 4]) = vh;
            *(v4bf*)(&dAl[arow[i] * AS + ac4 * 4]) = vl;
        }
        #pragma unroll
        for (int j = 0; j < 4; ++j) {
            float v4a[4] = {pb[j].x, pb[j].y, pb[j].z, pb[j].w};
            #pragma unroll
            for (int e = 0; e < 4; ++e) {
                bf16_t h, l;
                split_bf16(v4a[e], h, l);
                int n = bn4 * 4 + e;               // transpose: LDS is [n][k]
                dBh[(sg * BN + n) * BS + bk[j]] = h;
                dBl[(sg * BN + n) * BS + bk[j]] = l;
            }
        }
    };

    // pipeline prologue: fill buffer 0
    prefetch(0);
    stage(0);

    #pragma unroll 1
    for (int s = 0; s < NSTEP; ++s) {
        __syncthreads();   // buf[s&1] visible; all done computing buf[(s+1)&1]

        const int cur = s & 1;
        const bool more = (s + 1) < NSTEP;

        // issue global loads for step s+1 immediately (max latency cover)
        if (more) prefetch(s + 1);

        // ---- fragments + WMMA for step s (from buf[cur]) ----
        const bf16_t* cAh = sA_hi[cur];
        const bf16_t* cAl = sA_lo[cur];
        const bf16_t* cBh = sB_hi[cur];
        const bf16_t* cBl = sB_lo[cur];

        v16bf bh[4], bl[4];
        #pragma unroll
        for (int g = 0; g < 4; ++g) {
            bh[g] = ld_frag(&cBh[(g * BN + wc * 16 + lr) * BS + kb]);
            bl[g] = ld_frag(&cBl[(g * BN + wc * 16 + lr) * BS + kb]);
        }
        v16bf ah[2], al[2];
        #pragma unroll
        for (int mt = 0; mt < 2; ++mt) {
            ah[mt] = ld_frag(&cAh[(wr * 32 + mt * 16 + lr) * AS + kb]);
            al[mt] = ld_frag(&cAl[(wr * 32 + mt * 16 + lr) * AS + kb]);
        }
        #pragma unroll
        for (int g = 0; g < 4; ++g) {
            #pragma unroll
            for (int mt = 0; mt < 2; ++mt) {
                acc[g][mt] = __builtin_amdgcn_wmma_f32_16x16x32_bf16(
                    false, ah[mt], false, bh[g], (short)0, acc[g][mt], false, false);
                acc[g][mt] = __builtin_amdgcn_wmma_f32_16x16x32_bf16(
                    false, al[mt], false, bh[g], (short)0, acc[g][mt], false, false);
                acc[g][mt] = __builtin_amdgcn_wmma_f32_16x16x32_bf16(
                    false, ah[mt], false, bl[g], (short)0, acc[g][mt], false, false);
            }
        }

        // ---- convert+store step s+1 into the other buffer; overlaps WMMAs,
        //      no fence needed (all threads are past this iteration's barrier)
        if (more) stage(1 - cur);
    }

    // ---- fused pointwise epilogue (each wave owns z,i,f,o for its elems) ----
    const size_t BU  = (size_t)B_ * U_;
    const int    col = C0 + wc * 16 + lr;
    const float  bzv = pbz[col], biv = pbi[col], bfv = pbf[col], bov = pbo[col];

    #pragma unroll
    for (int mt = 0; mt < 2; ++mt) {
        const int rbase = R0 + wr * 32 + mt * 16 + hl * 8;  // C layout: M = v + 8*half
        #pragma unroll
        for (int v = 0; v < 8; ++v) {
            const int    r   = rbase + v;
            const size_t idx = (size_t)r * U_ + col;
            float pz = acc[0][mt][v] + bzv;
            float pi = acc[1][mt][v] + biv;
            float pf = acc[2][mt][v] + bfv;
            float po = acc[3][mt][v] + bov;
            float mp  = mprev[idx];
            float mtn = fmaxf(pi + mp, pi);
            float it  = __expf(pi - mtn);
            float ft  = __expf(pf + mp - mtn);
            float ot  = 1.0f / (1.0f + __expf(-po));
            float zt  = tanhf(pz);
            float ct  = ft * cprev[idx] + it * zt;
            float nt  = ft * nprev[idx] + it;
            float ht  = ot * (ct / (nt + 1e-8f));
            out[idx]            = ht;
            out[BU + idx]       = ct;
            out[2 * BU + idx]   = nt;
            out[3 * BU + idx]   = mtn;
        }
    }
}

extern "C" void kernel_launch(void* const* d_in, const int* in_sizes, int n_in,
                              void* d_out, int out_size, void* d_ws, size_t ws_size,
                              hipStream_t stream) {
    (void)in_sizes; (void)n_in; (void)out_size; (void)d_ws; (void)ws_size;
    const float* x     = (const float*)d_in[0];
    const float* hprev = (const float*)d_in[1];
    const float* cprev = (const float*)d_in[2];
    const float* nprev = (const float*)d_in[3];
    const float* mprev = (const float*)d_in[4];
    const float* Wz    = (const float*)d_in[5];
    const float* Wi    = (const float*)d_in[6];
    const float* Wf    = (const float*)d_in[7];
    const float* Wo    = (const float*)d_in[8];
    const float* bz    = (const float*)d_in[9];
    const float* bi    = (const float*)d_in[10];
    const float* bf    = (const float*)d_in[11];
    const float* bo    = (const float*)d_in[12];
    const float* Uz    = (const float*)d_in[13];
    const float* Ui    = (const float*)d_in[14];
    const float* Uf    = (const float*)d_in[15];
    const float* Uo    = (const float*)d_in[16];
    float* out = (float*)d_out;

    dim3 grid(U_ / BN, B_ / BM);   // (64, 2)
    dim3 block(256);               // 8 wave32s
    slstm_wmma_kernel<<<grid, block, 0, stream>>>(
        x, hprev, cprev, nprev, mprev,
        Wz, Wi, Wf, Wo, bz, bi, bf, bo,
        Uz, Ui, Uf, Uo, out);
}
